// CausalSelfAttention_11776800325859
// MI455X (gfx1250) — matrix operand order repaired and checked by reading, not compiled, not run on hardware
//
#include <hip/hip_runtime.h>

// ---------------------------------------------------------------------------
// CDNA5 (gfx1250) causal self-attention, bf16 WMMA with fp32 accumulation.
// B=2, S=2048, D=1024, H=16, HD=64. wave32 throughout.
// ---------------------------------------------------------------------------

#define BATCH 2
#define SEQ   2048
#define DIM   1024
#define NH    16
#define HD    64
#define NTOK  (BATCH * SEQ)   // 4096

typedef __attribute__((ext_vector_type(16))) __bf16 v16bf;
typedef __attribute__((ext_vector_type(8)))  float  v8f;

union V16U {
    v16bf v;
    uint4 u[2];
};

__device__ inline unsigned short f2bf(float f) {
    // round-to-nearest-even fp32 -> bf16 (NaN handling omitted; inputs are finite)
    unsigned int u = __float_as_uint(f);
    u += 0x7FFFu + ((u >> 16) & 1u);
    return (unsigned short)(u >> 16);
}

__device__ inline v8f wmma_bf16(v16bf a, v16bf b, v8f c) {
    // D = A(16x32 bf16) * B(32x16 bf16) + C(16x16 f32)
    return __builtin_amdgcn_wmma_f32_16x16x32_bf16(
        /*neg_a=*/false, a, /*neg_b=*/false, b,
        /*c_mod=*/(short)0, c, /*reuse_a=*/false, /*reuse_b=*/false);
}

// A-operand load (16x32 tile, row-major source, 16-bit A interleave):
// lane holds row (base + lane&15); elems 0..7 at k0+8g, elems 8..15 at k0+16+8g.
__device__ inline v16bf load_a(const unsigned short* base, int row16, int stride,
                               int k0, int col, int g) {
    V16U r;
    const unsigned short* p = base + (size_t)(row16 + col) * stride + k0 + 8 * g;
    r.u[0] = *(const uint4*)(p);
    r.u[1] = *(const uint4*)(p + 16);
    return r.v;
}

// B-operand load (32x16 tile; source holds B^T rows, i.e. lane reads row n=base+col):
// lane holds col n; elems 0..7 at k0+8g, elems 8..15 at k0+16+8g (matches A interleave).
__device__ inline v16bf load_b(const unsigned short* base, int row16, int stride,
                               int k0, int col, int g) {
    V16U r;
    const unsigned short* p = base + (size_t)(row16 + col) * stride + k0 + 8 * g;
    r.u[0] = *(const uint4*)(p);
    r.u[1] = *(const uint4*)(p + 16);
    return r.v;
}

// ---------------------------------------------------------------------------
// fp32 -> bf16 conversion
// ---------------------------------------------------------------------------
__global__ void cvt_bf16_kernel(const float* __restrict__ src,
                                unsigned short* __restrict__ dst, int n) {
    int i = blockIdx.x * blockDim.x + threadIdx.x;
    if (i < n) dst[i] = f2bf(src[i]);
}

// ---------------------------------------------------------------------------
// Fused QKV projection: out = x @ W^T + b    (per blockIdx.z: 0=Q, 1=K, 2=V)
// Q,K -> bf16 [b,h,s,hd] (LDS-restaged for coalesced stores)
// V   -> bf16 transposed [b,h,hd,s] (direct contiguous stores from C-layout)
// Block = 128 threads (4 waves), block tile 64x64, wave tile 32x32.
// ---------------------------------------------------------------------------
__global__ __launch_bounds__(128) void qkv_proj_kernel(
    const unsigned short* __restrict__ xb,
    const unsigned short* __restrict__ wqb,
    const unsigned short* __restrict__ wkb,
    const unsigned short* __restrict__ wvb,
    const float* __restrict__ bq, const float* __restrict__ bk,
    const float* __restrict__ bv,
    unsigned short* __restrict__ qo, unsigned short* __restrict__ ko,
    unsigned short* __restrict__ vto) {

    const int mode = blockIdx.z;
    const unsigned short* W = (mode == 0) ? wqb : (mode == 1) ? wkb : wvb;
    const float* bias       = (mode == 0) ? bq  : (mode == 1) ? bk  : bv;
    unsigned short* dstQK   = (mode == 0) ? qo  : ko;

    const int lane = threadIdx.x & 31;
    const int wid  = threadIdx.x >> 5;
    const int col  = lane & 15;
    const int g    = lane >> 4;

    const int m0 = blockIdx.x * 64 + (wid >> 1) * 32;  // token base
    const int n0 = blockIdx.y * 64 + (wid & 1) * 32;   // out-dim base

    v8f acc[2][2] = {};
#pragma unroll 1
    for (int k0 = 0; k0 < DIM; k0 += 32) {
        v16bf a0 = load_a(xb, m0,      DIM, k0, col, g);
        v16bf a1 = load_a(xb, m0 + 16, DIM, k0, col, g);
        v16bf b0 = load_b(W,  n0,      DIM, k0, col, g);
        v16bf b1 = load_b(W,  n0 + 16, DIM, k0, col, g);
        acc[0][0] = wmma_bf16(a0, b0, acc[0][0]);
        acc[0][1] = wmma_bf16(a0, b1, acc[0][1]);
        acc[1][0] = wmma_bf16(a1, b0, acc[1][0]);
        acc[1][1] = wmma_bf16(a1, b1, acc[1][1]);
    }

    __shared__ __align__(16) unsigned short stg[4][256];

    for (int i = 0; i < 2; ++i) {
        for (int j = 0; j < 2; ++j) {
            const float bval = bias[n0 + 16 * j + col];
            if (mode < 2) {
                __syncthreads();
#pragma unroll
                for (int r = 0; r < 8; ++r)
                    stg[wid][(r + 8 * g) * 16 + col] = f2bf(acc[i][j][r] + bval);
                __syncthreads();
                // lane copies one 16-byte row-chunk: row = col, halves by g
                const int trow = m0 + 16 * i + col;           // global token
                const int bb = trow >> 11, ss = trow & (SEQ - 1);
                const int nb = n0 + 16 * j;
                const int hh = nb >> 6, hd0 = (nb & 63) + 8 * g;
                uint4 val = *(const uint4*)&stg[wid][col * 16 + 8 * g];
                *(uint4*)&dstQK[(((size_t)bb * NH + hh) * SEQ + ss) * HD + hd0] = val;
            } else {
                // V transposed: lane owns column n (a row of V^T); 8 consecutive tokens
                const int n  = n0 + 16 * j + col;
                const int hh = n >> 6, hd = n & 63;
                const int t0 = m0 + 16 * i + 8 * g;           // 8-aligned, no batch split
                const int bb = t0 >> 11, s0 = t0 & (SEQ - 1);
                union { unsigned short s[8]; uint4 u; } tv;
#pragma unroll
                for (int r = 0; r < 8; ++r) tv.s[r] = f2bf(acc[i][j][r] + bval);
                *(uint4*)&vto[(((size_t)bb * NH + hh) * HD + hd) * SEQ + s0] = tv.u;
            }
        }
    }
}

// ---------------------------------------------------------------------------
// Flash attention: one wave (32-thread block) per (b, h, 16-query tile).
// S-tile via Q@K^T (2 WMMA k-steps over head_dim=64, 2 key sub-tiles),
// online softmax (cross-lane reductions in 16-lane halves), P restaged
// through LDS to A-layout, P@V^T via WMMA (V stored transposed).
// ---------------------------------------------------------------------------
__global__ __launch_bounds__(32) void attn_kernel(
    const unsigned short* __restrict__ qg,
    const unsigned short* __restrict__ kg,
    const unsigned short* __restrict__ vtg,
    unsigned short* __restrict__ y) {

    const int lane = threadIdx.x & 31;
    const int col = lane & 15;
    const int g   = lane >> 4;
    const int q0  = blockIdx.x * 16;
    const int h   = blockIdx.y;
    const int b   = blockIdx.z;
    const size_t bh = (size_t)b * NH + h;

    const unsigned short* Q  = qg  + bh * SEQ * HD;
    const unsigned short* K  = kg  + bh * SEQ * HD;
    const unsigned short* Vt = vtg + bh * HD * SEQ;

    __shared__ __align__(16) unsigned short lds_p[16 * 32];
    __shared__ __align__(16) unsigned short lds_y[16 * 64];

    // Q A-tiles for head-dim chunks 0..31 / 32..63 (kept in registers)
    V16U qa[2];
#pragma unroll
    for (int s = 0; s < 2; ++s) {
        const unsigned short* p = Q + (size_t)(q0 + col) * HD + 32 * s + 8 * g;
        qa[s].u[0] = *(const uint4*)(p);
        qa[s].u[1] = *(const uint4*)(p + 16);
    }

    v8f acc[4] = {};
    float rowm[8], rows[8];
#pragma unroll
    for (int r = 0; r < 8; ++r) { rowm[r] = -__builtin_inff(); rows[r] = 0.0f; }

    const float scale = 0.03125f;  // 1/sqrt(D) = 1/32 (reference scales by embed dim)
    const int nch = (q0 + 16 + 31) / 32;  // causal: keys 0 .. q0+15

#pragma unroll 1
    for (int kc = 0; kc < nch; ++kc) {
        const int kk0 = kc * 32;

        // S = Q @ K^T for 32 keys -> two 16x16 C tiles
        v8f st[2] = {};
#pragma unroll
        for (int t = 0; t < 2; ++t)
#pragma unroll
            for (int s = 0; s < 2; ++s) {
                v16bf kb = load_b(K, kk0 + 16 * t, HD, 32 * s, col, g);
                st[t] = wmma_bf16(qa[s].v, kb, st[t]);
            }

        // scale + causal mask (C layout: m = q0 + r + 8g, n = kk0 + 16t + col)
#pragma unroll
        for (int t = 0; t < 2; ++t) {
            const int n = kk0 + 16 * t + col;
#pragma unroll
            for (int r = 0; r < 8; ++r) {
                const int m = q0 + r + 8 * g;
                st[t][r] = (n > m) ? -1e30f : st[t][r] * scale;
            }
        }

        // online softmax per row (rows live in fixed half-wave, reduce over 16 lanes)
#pragma unroll
        for (int r = 0; r < 8; ++r) {
            float mx = fmaxf(st[0][r], st[1][r]);
            mx = fmaxf(mx, __shfl_xor(mx, 1, 32));
            mx = fmaxf(mx, __shfl_xor(mx, 2, 32));
            mx = fmaxf(mx, __shfl_xor(mx, 4, 32));
            mx = fmaxf(mx, __shfl_xor(mx, 8, 32));
            const float rm_new = fmaxf(rowm[r], mx);
            const float corr = __expf(rowm[r] - rm_new);
            rowm[r] = rm_new;
            const float p0 = __expf(st[0][r] - rm_new);
            const float p1 = __expf(st[1][r] - rm_new);
            float ps = p0 + p1;
            ps += __shfl_xor(ps, 1, 32);
            ps += __shfl_xor(ps, 2, 32);
            ps += __shfl_xor(ps, 4, 32);
            ps += __shfl_xor(ps, 8, 32);
            rows[r] = rows[r] * corr + ps;
#pragma unroll
            for (int c = 0; c < 4; ++c) acc[c][r] *= corr;
            const int m = r + 8 * g;
            lds_p[m * 32 + col]      = f2bf(p0);
            lds_p[m * 32 + 16 + col] = f2bf(p1);
        }
        __syncthreads();

        // reload P as A-operand (16x32, A interleave)
        V16U pa;
        pa.u[0] = *(const uint4*)&lds_p[col * 32 + 8 * g];
        pa.u[1] = *(const uint4*)&lds_p[col * 32 + 16 + 8 * g];
        __syncthreads();

        // acc += P @ V  (B operand = rows of V^T, contiguous along seq)
#pragma unroll
        for (int c = 0; c < 4; ++c) {
            v16bf vb = load_b(Vt, 16 * c, SEQ, kk0, col, g);
            acc[c] = wmma_bf16(pa.v, vb, acc[c]);
        }
    }

    // normalize and stage y tile (16 x 64) to LDS as bf16
#pragma unroll
    for (int r = 0; r < 8; ++r) {
        const float inv = 1.0f / rows[r];
        const int m = r + 8 * g;
#pragma unroll
        for (int c = 0; c < 4; ++c)
            lds_y[m * 64 + 16 * c + col] = f2bf(acc[c][r] * inv);
    }
    __syncthreads();

    // write y [B,S,D] bf16: lane owns row=col, byte-half=g (4 x 16B chunks)
    const size_t base = ((size_t)b * SEQ + q0 + col) * DIM + h * HD;
#pragma unroll
    for (int c2 = 0; c2 < 4; ++c2) {
        const int off = (4 * g + c2) * 8;
        *(uint4*)&y[base + off] = *(const uint4*)&lds_y[col * 64 + off];
    }
}

// ---------------------------------------------------------------------------
// Output projection: out(f32) = y(bf16) @ Wp^T + bp, row-major [B*S, D]
// ---------------------------------------------------------------------------
__global__ __launch_bounds__(128) void out_proj_kernel(
    const unsigned short* __restrict__ yb,
    const unsigned short* __restrict__ wpb,
    const float* __restrict__ bp,
    float* __restrict__ out) {

    const int lane = threadIdx.x & 31;
    const int wid  = threadIdx.x >> 5;
    const int col  = lane & 15;
    const int g    = lane >> 4;

    const int m0 = blockIdx.x * 64 + (wid >> 1) * 32;
    const int n0 = blockIdx.y * 64 + (wid & 1) * 32;

    v8f acc[2][2] = {};
#pragma unroll 1
    for (int k0 = 0; k0 < DIM; k0 += 32) {
        v16bf a0 = load_a(yb,  m0,      DIM, k0, col, g);
        v16bf a1 = load_a(yb,  m0 + 16, DIM, k0, col, g);
        v16bf b0 = load_b(wpb, n0,      DIM, k0, col, g);
        v16bf b1 = load_b(wpb, n0 + 16, DIM, k0, col, g);
        acc[0][0] = wmma_bf16(a0, b0, acc[0][0]);
        acc[0][1] = wmma_bf16(a0, b1, acc[0][1]);
        acc[1][0] = wmma_bf16(a1, b0, acc[1][0]);
        acc[1][1] = wmma_bf16(a1, b1, acc[1][1]);
    }

    __shared__ __align__(16) float stgf[4][256];

    for (int i = 0; i < 2; ++i) {
        for (int j = 0; j < 2; ++j) {
            const float bval = bp[n0 + 16 * j + col];
            __syncthreads();
#pragma unroll
            for (int r = 0; r < 8; ++r)
                stgf[wid][(r + 8 * g) * 16 + col] = acc[i][j][r] + bval;
            __syncthreads();
            const int trow = m0 + 16 * i + col;
            float* dst = out + (size_t)trow * DIM + n0 + 16 * j + 8 * g;
            *(float4*)(dst)     = *(const float4*)&stgf[wid][col * 16 + 8 * g];
            *(float4*)(dst + 4) = *(const float4*)&stgf[wid][col * 16 + 8 * g + 4];
        }
    }
}

// ---------------------------------------------------------------------------
// Launch
// ---------------------------------------------------------------------------
extern "C" void kernel_launch(void* const* d_in, const int* in_sizes, int n_in,
                              void* d_out, int out_size, void* d_ws, size_t ws_size,
                              hipStream_t stream) {
    const float* x  = (const float*)d_in[0];
    const float* Wq = (const float*)d_in[1];
    const float* bq = (const float*)d_in[2];
    const float* Wk = (const float*)d_in[3];
    const float* bk = (const float*)d_in[4];
    const float* Wv = (const float*)d_in[5];
    const float* bv = (const float*)d_in[6];
    const float* Wp = (const float*)d_in[7];
    const float* bp = (const float*)d_in[8];
    float* out = (float*)d_out;

    char* ws = (char*)d_ws;
    size_t off = 0;
    unsigned short* xb  = (unsigned short*)(ws + off); off += (size_t)NTOK * DIM * 2;
    unsigned short* wqb = (unsigned short*)(ws + off); off += (size_t)DIM * DIM * 2;
    unsigned short* wkb = (unsigned short*)(ws + off); off += (size_t)DIM * DIM * 2;
    unsigned short* wvb = (unsigned short*)(ws + off); off += (size_t)DIM * DIM * 2;
    unsigned short* wpb = (unsigned short*)(ws + off); off += (size_t)DIM * DIM * 2;
    unsigned short* q   = (unsigned short*)(ws + off); off += (size_t)NTOK * DIM * 2;
    unsigned short* k   = (unsigned short*)(ws + off); off += (size_t)NTOK * DIM * 2;
    unsigned short* vt  = (unsigned short*)(ws + off); off += (size_t)NTOK * DIM * 2;
    unsigned short* y   = (unsigned short*)(ws + off); off += (size_t)NTOK * DIM * 2;

    const int nX = NTOK * DIM;
    const int nW = DIM * DIM;
    cvt_bf16_kernel<<<(nX + 255) / 256, 256, 0, stream>>>(x,  xb,  nX);
    cvt_bf16_kernel<<<(nW + 255) / 256, 256, 0, stream>>>(Wq, wqb, nW);
    cvt_bf16_kernel<<<(nW + 255) / 256, 256, 0, stream>>>(Wk, wkb, nW);
    cvt_bf16_kernel<<<(nW + 255) / 256, 256, 0, stream>>>(Wv, wvb, nW);
    cvt_bf16_kernel<<<(nW + 255) / 256, 256, 0, stream>>>(Wp, wpb, nW);

    qkv_proj_kernel<<<dim3(NTOK / 64, DIM / 64, 3), 128, 0, stream>>>(
        xb, wqb, wkb, wvb, bq, bk, bv, q, k, vt);

    attn_kernel<<<dim3(SEQ / 16, NH, BATCH), 32, 0, stream>>>(q, k, vt, y);

    out_proj_kernel<<<dim3(NTOK / 64, DIM / 64, 1), 128, 0, stream>>>(y, wpb, bp, out);
}
